// Head_2705829396816
// MI455X (gfx1250) — matrix operand order repaired and checked by reading, not compiled, not run on hardware
//
#include <hip/hip_runtime.h>
#include <hip/hip_bf16.h>

typedef __attribute__((ext_vector_type(16))) _Float16     v16h;
typedef __attribute__((ext_vector_type(8)))  _Float16     v8h;
typedef __attribute__((ext_vector_type(8)))  float        v8f;
typedef __attribute__((ext_vector_type(4)))  unsigned int u32x4;
typedef __attribute__((ext_vector_type(8)))  int          i32x8;
typedef __attribute__((ext_vector_type(4)))  int          i32x4;

#define BATCH 8
#define SEQ   2048
#define EMBD  1024
#define HDIM  64

static __device__ __forceinline__ v16h cat8(v8h lo, v8h hi) {
    return __builtin_shufflevector(lo, hi, 0, 1, 2, 3, 4, 5, 6, 7,
                                           8, 9, 10, 11, 12, 13, 14, 15);
}

// ---------------------------------------------------------------------------
// TDM: issue a 2-D tensor tile load (global -> LDS) via the Tensor Data Mover.
// Descriptor per CDNA5 ISA 8.3/8.4:  group0 = {count, lds_addr, global_addr,
// type=2};  group1 = {data_size, tensor dims (huge, no OOB clip), tile dims,
// dim0 stride}.  Tracked by TENSORcnt.  (6-arg clang-23 builtin form.)
// ---------------------------------------------------------------------------
static __device__ __forceinline__ void tdm_load_tile(unsigned lds_addr,
                                                     const void* gptr,
                                                     int data_size_code,
                                                     int tile0, int tile1,
                                                     long long stride0_elems)
{
    const unsigned long long ga = (unsigned long long)(uintptr_t)gptr;
    u32x4 g0 = { 1u,                                   // count = 1 descriptor
                 lds_addr,                             // LDS byte address
                 (unsigned)(ga & 0xffffffffu),
                 (unsigned)((ga >> 32) & 0x1ffffffu) | (2u << 30) };  // type=2

    const unsigned dim = 1u << 20;                     // tensor_dim0/1: huge
    i32x8 g1;
    g1[0] = data_size_code << 16;                      // wg_mask=0, data_size
    g1[1] = (int)((dim & 0xffffu) << 16);              // tensor_dim0[15:0]
    g1[2] = (int)(((dim >> 16) & 0xffffu)              // tensor_dim0[31:16]
                | ((dim & 0xffffu) << 16));            // tensor_dim1[15:0]
    g1[3] = (int)(((dim >> 16) & 0xffffu)              // tensor_dim1[31:16]
                | ((unsigned)tile0 << 16));            // tile_dim0
    g1[4] = tile1 & 0xffff;                            // tile_dim1 (tile2 = 0)
    g1[5] = (int)(stride0_elems & 0xffffffffll);       // dim0_stride[31:0]
    g1[6] = (int)(((stride0_elems >> 32) & 0xffffll)   // dim0_stride[47:32]
                | ((stride0_elems & 0xffffll) << 16)); // dim1_stride[15:0]
    g1[7] = (int)((stride0_elems >> 16) & 0xffffffffll);
    i32x4 z4 = { 0, 0, 0, 0 };
    i32x8 z8 = { 0, 0, 0, 0, 0, 0, 0, 0 };
    __builtin_amdgcn_tensor_load_to_lds(g0, g1, z4, z4, z8, 0);
}

// ---------------------------------------------------------------------------
// Kernel 1: QKV projection -> f16 workspace.
// ---------------------------------------------------------------------------
__global__ void __launch_bounds__(128)
qkv_proj_kernel(const float* __restrict__ x,
                const float* __restrict__ Wq,
                const float* __restrict__ Wk,
                const float* __restrict__ Wv,
                _Float16* __restrict__ qh,
                _Float16* __restrict__ kh,
                _Float16* __restrict__ vt)
{
    const int lane = threadIdx.x & 31;
    const int wave = threadIdx.x >> 5;
    const bool lo16 = lane < 16;
    const int r = lane & 15;

    const int rowBase = (blockIdx.x * 4 + wave) * 16;
    const int y = blockIdx.y;
    const float* __restrict__ W = (y == 0) ? Wq : (y == 1) ? Wk : Wv;

    const int aRow = rowBase + r;

    v8f acc[4] = {};

    for (int kb = 0; kb < EMBD; kb += 32) {
        v16h a;
        const float* ax = x + (size_t)aRow * EMBD + kb + (lo16 ? 0 : 16);
#pragma unroll
        for (int i = 0; i < 8; ++i) a[i]     = (_Float16)ax[i];
#pragma unroll
        for (int i = 0; i < 8; ++i) a[8 + i] = (_Float16)ax[8 + i];

#pragma unroll
        for (int nt = 0; nt < 4; ++nt) {
            v16h bfrag;
            const float* bw = W + (size_t)(kb + (lo16 ? 0 : 16)) * HDIM + nt * 16 + r;
#pragma unroll
            for (int i = 0; i < 16; ++i) bfrag[i] = (_Float16)bw[i * HDIM];
            acc[nt] = __builtin_amdgcn_wmma_f32_16x16x32_f16(
                false, a, false, bfrag, (short)0, acc[nt], false, false);
        }
    }

    _Float16* __restrict__ outh = (y == 0) ? qh : kh;
#pragma unroll
    for (int rr = 0; rr < 8; ++rr) {
        const int m = rr + (lo16 ? 0 : 8);
        const int row = rowBase + m;
#pragma unroll
        for (int nt = 0; nt < 4; ++nt) {
            const int h = nt * 16 + r;
            const _Float16 hv = (_Float16)acc[nt][rr];
            if (y == 2) {
                const int b = row >> 11;
                const int t = row & (SEQ - 1);
                vt[((size_t)b * HDIM + h) * SEQ + t] = hv;
            } else {
                outh[(size_t)row * HDIM + h] = hv;
            }
        }
    }
}

// ---------------------------------------------------------------------------
// Kernel 2: causal flash attention.  grid = (T/64, B), block = 128 (4 waves).
// Per 32-key chunk: 4 WMMAs for S = Q*K^T (K^T via contiguous b128 global
// loads + prefetch), online softmax through per-wave LDS, and O += P*V with
// the V^T tile DMA'd into LDS by the Tensor Data Mover, double-buffered on
// TENSORcnt so chunk j+1 streams in while chunk j computes.
// ---------------------------------------------------------------------------
__global__ void __launch_bounds__(128)
attn_fwd_kernel(const _Float16* __restrict__ qh,
                const _Float16* __restrict__ kh,
                const _Float16* __restrict__ vt,
                float* __restrict__ out)
{
    __shared__ __align__(16) _Float16 vBuf[4][2][HDIM * 32];  // TDM double buffer
    __shared__ __align__(16) float    sS[4][16][32];
    __shared__ __align__(16) _Float16 sP[4][16][32];
    __shared__ float sM[4][16];
    __shared__ float sL[4][16];
    __shared__ float sA[4][16];

    const int lane = threadIdx.x & 31;
    const int w    = threadIdx.x >> 5;
    const bool lo16 = lane < 16;
    const int r = lane & 15;

    const int b     = blockIdx.y;
    const int qBase = (blockIdx.x * 4 + w) * 16;

    const _Float16* __restrict__ Qb = qh + (size_t)b * SEQ * HDIM;
    const _Float16* __restrict__ Kb = kh + (size_t)b * SEQ * HDIM;
    const _Float16* __restrict__ Vb = vt + (size_t)b * HDIM * SEQ;   // [h][t]

    // ---- Q tile: two 16x32 A-fragments, pure b128 loads
    v16h qa[2];
#pragma unroll
    for (int cc = 0; cc < 2; ++cc) {
        const _Float16* qp = Qb + (size_t)(qBase + r) * HDIM + cc * 32 + (lo16 ? 0 : 8);
        qa[cc] = cat8(*(const v8h*)qp, *(const v8h*)(qp + 16));
    }

    if (lo16) { sM[w][r] = -1e30f; sL[w][r] = 0.0f; }
    asm volatile("s_wait_dscnt 0" ::: "memory");

    v8f o0 = {}, o1 = {}, o2 = {}, o3 = {};

    const int nChunks = (qBase + 15) / 32 + 1;   // causal: keys 0 .. qBase+15

    // ---- prologue: TDM-stream chunk 0's V^T tile (32 keys x 64 h) into LDS
    tdm_load_tile((unsigned)(uintptr_t)&vBuf[w][0][0], Vb,
                  /*data_size=2B*/ 1, /*tile0=*/32, /*tile1=*/HDIM,
                  /*stride0=*/SEQ);

    for (int j = 0; j < nChunks; ++j) {
        const int kbase = j * 32;
        const int buf   = j & 1;

        // ---- stream next chunk's V^T tile; keep exactly one DMA in flight
        if (j + 1 < nChunks) {
            tdm_load_tile((unsigned)(uintptr_t)&vBuf[w][buf ^ 1][0],
                          Vb + kbase + 32, 1, 32, HDIM, SEQ);
            __builtin_prefetch(Kb + (size_t)(kbase + 32 + r) * HDIM, 0, 0);
            __builtin_prefetch(Kb + (size_t)(kbase + 48 + r) * HDIM, 0, 0);
            __builtin_amdgcn_s_wait_tensorcnt(1);
        } else {
            __builtin_amdgcn_s_wait_tensorcnt(0);
        }
        asm volatile("" ::: "memory");

        // ---------- S = Q * K^T : two 16-key subtiles, K-dim 64 ----------
        v8f s01[2] = {};
#pragma unroll
        for (int st = 0; st < 2; ++st) {
            const int keyRow = kbase + st * 16 + r;
#pragma unroll
            for (int cc = 0; cc < 2; ++cc) {
                const _Float16* kp = Kb + (size_t)keyRow * HDIM + cc * 32 + (lo16 ? 0 : 8);
                v16h bk = cat8(*(const v8h*)kp, *(const v8h*)(kp + 16));
                s01[st] = __builtin_amdgcn_wmma_f32_16x16x32_f16(
                    false, qa[cc], false, bk, (short)0, s01[st], false, false);
            }
        }

        // ---------- scale + causal mask, spill S tile to LDS ----------
#pragma unroll
        for (int rr = 0; rr < 8; ++rr) {
            const int row = rr + (lo16 ? 0 : 8);
            const int qidx = qBase + row;
#pragma unroll
            for (int st = 0; st < 2; ++st) {
                const int key = kbase + st * 16 + r;
                float val = s01[st][rr] * 0.125f;        // 1/sqrt(64)
                if (key > qidx) val = -1e30f;
                sS[w][row][st * 16 + r] = val;
            }
        }
        asm volatile("s_wait_dscnt 0" ::: "memory");

        // ---------- online softmax per row (lanes 0-15 own one row) ----------
        if (lo16) {
            const int row = r;
            float mOld = sM[w][row];
            float mNew = mOld;
#pragma unroll
            for (int c = 0; c < 32; ++c) mNew = fmaxf(mNew, sS[w][row][c]);
            const float alpha = __expf(mOld - mNew);
            float lsum = 0.0f;
#pragma unroll
            for (int c = 0; c < 32; ++c) {
                const float p = __expf(sS[w][row][c] - mNew);
                lsum += p;
                sP[w][row][c] = (_Float16)p;
            }
            sA[w][row] = alpha;
            sM[w][row] = mNew;
            sL[w][row] = alpha * sL[w][row] + lsum;
        }
        asm volatile("s_wait_dscnt 0" ::: "memory");

        // ---------- rescale O by alpha (per C-layout row) ----------
        float a8[8];
#pragma unroll
        for (int rr = 0; rr < 8; ++rr) a8[rr] = sA[w][rr + (lo16 ? 0 : 8)];
#pragma unroll
        for (int rr = 0; rr < 8; ++rr) {
            o0[rr] *= a8[rr]; o1[rr] *= a8[rr]; o2[rr] *= a8[rr]; o3[rr] *= a8[rr];
        }

        // ---------- reload P as an A-fragment (two ds_load_b128) ----------
        v16h pa;
        {
            const _Float16* pp = &sP[w][r][lo16 ? 0 : 8];
            pa = cat8(*(const v8h*)pp, *(const v8h*)(pp + 16));
        }

        // ---------- O += P * V : V^T fragments from the TDM-staged LDS tile --
#pragma unroll
        for (int nt = 0; nt < 4; ++nt) {
            const _Float16* vp = &vBuf[w][buf][(nt * 16 + r) * 32 + (lo16 ? 0 : 8)];
            v16h bv = cat8(*(const v8h*)vp, *(const v8h*)(vp + 16));
            if      (nt == 0) o0 = __builtin_amdgcn_wmma_f32_16x16x32_f16(false, pa, false, bv, (short)0, o0, false, false);
            else if (nt == 1) o1 = __builtin_amdgcn_wmma_f32_16x16x32_f16(false, pa, false, bv, (short)0, o1, false, false);
            else if (nt == 2) o2 = __builtin_amdgcn_wmma_f32_16x16x32_f16(false, pa, false, bv, (short)0, o2, false, false);
            else              o3 = __builtin_amdgcn_wmma_f32_16x16x32_f16(false, pa, false, bv, (short)0, o3, false, false);
        }
    }

    // ---------- epilogue: divide by row sum, write (B,T,H) fp32 ----------
    asm volatile("s_wait_dscnt 0" ::: "memory");
    float l8[8];
#pragma unroll
    for (int rr = 0; rr < 8; ++rr) l8[rr] = sL[w][rr + (lo16 ? 0 : 8)];

#pragma unroll
    for (int rr = 0; rr < 8; ++rr) {
        const int row = rr + (lo16 ? 0 : 8);
        float* op = out + ((size_t)b * SEQ + qBase + row) * HDIM + r;
        const float inv = 1.0f / l8[rr];
        op[0]  = o0[rr] * inv;
        op[16] = o1[rr] * inv;
        op[32] = o2[rr] * inv;
        op[48] = o3[rr] * inv;
    }
}

// ---------------------------------------------------------------------------
extern "C" void kernel_launch(void* const* d_in, const int* in_sizes, int n_in,
                              void* d_out, int out_size, void* d_ws, size_t ws_size,
                              hipStream_t stream) {
    const float* x  = (const float*)d_in[0];
    const float* Wq = (const float*)d_in[1];
    const float* Wk = (const float*)d_in[2];
    const float* Wv = (const float*)d_in[3];

    _Float16* qh = (_Float16*)d_ws;
    _Float16* kh = qh + (size_t)BATCH * SEQ * HDIM;
    _Float16* vh = kh + (size_t)BATCH * SEQ * HDIM;   // transposed V

    qkv_proj_kernel<<<dim3((BATCH * SEQ) / 64, 3), 128, 0, stream>>>(
        x, Wq, Wk, Wv, qh, kh, vh);

    attn_fwd_kernel<<<dim3(SEQ / 64, BATCH), 128, 0, stream>>>(
        qh, kh, vh, (float*)d_out);
}
